// CpuLstmModel_15530601743000
// MI455X (gfx1250) — compile-verified, matrix operand order checked by reading, not run on hardware
//
#include <hip/hip_runtime.h>
#include <hip/hip_bf16.h>

typedef __bf16 bf16_t;
typedef __attribute__((ext_vector_type(16))) __bf16 v16bf;
typedef __attribute__((ext_vector_type(8)))  __bf16 v8bf;
typedef __attribute__((ext_vector_type(8)))  float  v8f;
typedef int v4i __attribute__((vector_size(16)));

#define NTT   365
#define NGRID 4096
#define NXF   32
#define HID   512
#define G4    (4*HID)   // 2048 gate rows
#define KDIM  (2*HID)   // 1024 = [x0 | h]

union V16 { v16bf v; v8bf h[2]; };

__device__ __forceinline__ float sigf(float x) { return 1.0f / (1.0f + __expf(-x)); }

// ---------------- CDNA5 async global->LDS staging (ASYNCcnt path), with safe fallback --------
#if __has_builtin(__builtin_amdgcn_global_load_async_to_lds_b128)
#define HAVE_ASYNC_LDS 1
#else
#define HAVE_ASYNC_LDS 0
#endif

__device__ __forceinline__ void copy_b128_to_lds(bf16_t* lds_dst, const bf16_t* gsrc) {
#if HAVE_ASYNC_LDS
    __builtin_amdgcn_global_load_async_to_lds_b128(
        (__attribute__((address_space(1))) v4i*)gsrc,
        (__attribute__((address_space(3))) v4i*)lds_dst, 0, 0);
#else
    *(uint4*)lds_dst = *(const uint4*)gsrc;
#endif
}

#if HAVE_ASYNC_LDS
#if __has_builtin(__builtin_amdgcn_s_wait_asynccnt)
#define WAIT_ASYNC(n) __builtin_amdgcn_s_wait_asynccnt(n)
#else
#define WAIT_ASYNC(n) asm volatile("s_wait_asynccnt %0" :: "i"(n) : "memory")
#endif
#else
#define WAIT_ASYNC(n) ((void)0)
#endif

// ---------------- init: pack [w_ih | w_hh] -> bf16 Wcat[2048][1024], bias sum, zero state ----
__global__ void lstm_init_kernel(const float* __restrict__ w_ih, const float* __restrict__ w_hh,
                                 const float* __restrict__ b_ih, const float* __restrict__ b_hh,
                                 bf16_t* __restrict__ Wcat, float* __restrict__ biassum,
                                 float* __restrict__ c, bf16_t* __restrict__ h0,
                                 bf16_t* __restrict__ h1) {
    const int tid = blockIdx.x * blockDim.x + threadIdx.x;
    const int nth = gridDim.x * blockDim.x;
    for (int i = tid; i < G4 * KDIM; i += nth) {
        const int r = i / KDIM, k = i % KDIM;
        const float w = (k < HID) ? w_ih[r * HID + k] : w_hh[r * HID + (k - HID)];
        Wcat[i] = (bf16_t)w;
    }
    for (int i = tid; i < G4; i += nth) biassum[i] = b_ih[i] + b_hh[i];
    for (int i = tid; i < NGRID * HID; i += nth) {
        c[i] = 0.0f;
        h0[i] = (bf16_t)0.0f;
        h1[i] = (bf16_t)0.0f;
    }
}

// ---------------- per-step input projection: x0 = relu(x @ W_in^T + b_in) -> bf16 -------------
__global__ void xproj_kernel(const float* __restrict__ x, const float* __restrict__ W_in,
                             const float* __restrict__ b_in, bf16_t* __restrict__ x0,
                             float* __restrict__ y, const float* __restrict__ b_out) {
    const int tid = blockIdx.x * blockDim.x + threadIdx.x;  // 0 .. 4096*512-1
    const int m = tid >> 9;
    const int j = tid & (HID - 1);
    const float* xr = x + (size_t)m * NXF;
    const float* wr = W_in + (size_t)j * NXF;
    float s = b_in[j];
#pragma unroll
    for (int k = 0; k < NXF; ++k) s += xr[k] * wr[k];
    s = s > 0.0f ? s : 0.0f;
    x0[tid] = (bf16_t)s;
    if (j == 0) y[m] = b_out[0];   // seed output with bias; gate blocks atomically add partials
}

// ---------------- staging helper: one K-chunk (A: 128x64, B: 256x64) into an LDS buffer ------
__device__ __forceinline__ void stage_chunk(bf16_t* As, bf16_t* Bs,
                                            const bf16_t* __restrict__ x0,
                                            const bf16_t* __restrict__ hprev,
                                            const bf16_t* __restrict__ Wcat,
                                            int m0, int nb, int tid, int chunk) {
    const bf16_t* asrc = (chunk < 8) ? x0 : hprev;
    const int kc = (chunk & 7) * 64;
#pragma unroll
    for (int i = 0; i < 4; ++i) {               // A: 1024 x 16B, 4 per thread
        const int idx = tid + i * 256;
        const int row = idx >> 3;
        const int c8  = idx & 7;
        copy_b128_to_lds(As + row * 64 + c8 * 8,
                         asrc + (size_t)(m0 + row) * HID + kc + c8 * 8);
    }
#pragma unroll
    for (int i = 0; i < 8; ++i) {               // B: 2048 x 16B, 8 per thread
        const int idx = tid + i * 256;
        const int row = idx >> 3;
        const int c8  = idx & 7;
        const int gi  = row >> 6;
        const int jj  = row & 63;
        copy_b128_to_lds(Bs + row * 64 + c8 * 8,
                         Wcat + (size_t)(gi * HID + nb + jj) * KDIM + chunk * 64 + c8 * 8);
    }
}

// ---------------- per-step fused gate GEMM (WMMA bf16) + LSTM cell + output projection -------
// grid = (4096/128, 512/64), block = 256 threads = 8 waves.
// Block tile: 128 M rows x 64 hidden cols, computed for all 4 gates (i,f,g,o).
__global__ void __launch_bounds__(256, 1)
lstm_step_kernel(const bf16_t* __restrict__ x0, const bf16_t* __restrict__ hprev,
                 bf16_t* __restrict__ hnext, const bf16_t* __restrict__ Wcat,
                 const float* __restrict__ biassum, float* __restrict__ c,
                 const float* __restrict__ W_out, float* __restrict__ y) {
    __shared__ __align__(16) bf16_t As[2][128 * 64];   // 2 x 16 KB double-buffered A chunk
    __shared__ __align__(16) bf16_t Bs[2][256 * 64];   // 2 x 32 KB double-buffered B chunk

    const int m0   = blockIdx.x * 128;
    const int nb   = blockIdx.y * 64;
    const int tid  = threadIdx.x;
    const int wave = tid >> 5;
    const int lane = tid & 31;
    const int lrow  = lane & 15;   // row (A) / col (B,C) index within 16
    const int lhalf = lane >> 4;   // K-half select (A/B), M-half select (C/D)
    const int klo   = lhalf * 8;

    v8f acc[4][4] = {};            // [gate][n-subtile], 128 VGPRs of f32 accumulators

    // prologue: stage chunk 0 into buffer 0 (12 async b128 per thread)
    stage_chunk(As[0], Bs[0], x0, hprev, Wcat, m0, nb, tid, 0);

    for (int chunk = 0; chunk < 16; ++chunk) {
        const int cur = chunk & 1;
        if (chunk < 15) {
            // stage next chunk into the other buffer, then wait for this chunk's 12 loads
            stage_chunk(As[cur ^ 1], Bs[cur ^ 1], x0, hprev, Wcat, m0, nb, tid, chunk + 1);
            WAIT_ASYNC(12);
        } else {
            WAIT_ASYNC(0);
        }
        __syncthreads();           // all waves' chunk data resident in LDS

        const bf16_t* Ab = As[cur];
        const bf16_t* Bb = Bs[cur];
#pragma unroll
        for (int ks = 0; ks < 64; ks += 32) {
            // A fragment: row M = lrow of this wave's 16-row slice; K pattern {klo..+7, klo+16..+23}
            V16 a;
            const bf16_t* ab = Ab + (wave * 16 + lrow) * 64 + ks + klo;
            a.h[0] = *(const v8bf*)(ab);
            a.h[1] = *(const v8bf*)(ab + 16);
#pragma unroll
            for (int g = 0; g < 4; ++g) {
#pragma unroll
                for (int n = 0; n < 4; ++n) {
                    V16 b;
                    const bf16_t* bb = Bb + (g * 64 + n * 16 + lrow) * 64 + ks + klo;
                    b.h[0] = *(const v8bf*)(bb);
                    b.h[1] = *(const v8bf*)(bb + 16);
                    acc[g][n] = __builtin_amdgcn_wmma_f32_16x16x32_bf16(
                        false, a.v, false, b.v, (short)0, acc[g][n], false, false);
                }
            }
        }
        __syncthreads();           // done reading buf[cur]; it may be overwritten at chunk+2
    }

    // ---- fused LSTM cell + partial output projection ----
    // C/D layout: lane holds column n = lrow; VGPR r holds row m = 8*lhalf + r.
    const int mwb = m0 + wave * 16 + lhalf * 8;
    float ypart[8];
#pragma unroll
    for (int r = 0; r < 8; ++r) ypart[r] = 0.0f;

#pragma unroll
    for (int n = 0; n < 4; ++n) {
        const int j = nb + n * 16 + lrow;          // hidden unit index
        const float bi  = biassum[0 * HID + j];
        const float bff = biassum[1 * HID + j];
        const float bg  = biassum[2 * HID + j];
        const float bo  = biassum[3 * HID + j];
        const float wout = W_out[j];
#pragma unroll
        for (int r = 0; r < 8; ++r) {
            const int m = mwb + r;
            const float iv = sigf(acc[0][n][r] + bi);
            const float fv = sigf(acc[1][n][r] + bff);
            const float gv = tanhf(acc[2][n][r] + bg);
            const float ov = sigf(acc[3][n][r] + bo);
            const size_t off = (size_t)m * HID + j;
            const float cn = fv * c[off] + iv * gv;
            const float hn = ov * tanhf(cn);
            c[off] = cn;
            hnext[off] = (bf16_t)hn;
            ypart[r] += hn * wout;
        }
    }
    // reduce across the 16 lanes sharing the same m (xor masks <16 stay within halves)
#pragma unroll
    for (int r = 0; r < 8; ++r) {
        float v = ypart[r];
        v += __shfl_xor(v, 1);
        v += __shfl_xor(v, 2);
        v += __shfl_xor(v, 4);
        v += __shfl_xor(v, 8);
        if (lrow == 0) atomicAdd(&y[mwb + r], v);
    }
}

extern "C" void kernel_launch(void* const* d_in, const int* in_sizes, int n_in,
                              void* d_out, int out_size, void* d_ws, size_t ws_size,
                              hipStream_t stream) {
    const float* inputs = (const float*)d_in[0];  // [365][4096][32]
    const float* W_in   = (const float*)d_in[1];  // [512][32]
    const float* b_in   = (const float*)d_in[2];  // [512]
    const float* w_ih   = (const float*)d_in[3];  // [2048][512]
    const float* w_hh   = (const float*)d_in[4];  // [2048][512]
    const float* b_ih   = (const float*)d_in[5];  // [2048]
    const float* b_hh   = (const float*)d_in[6];  // [2048]
    const float* W_out  = (const float*)d_in[7];  // [1][512]
    const float* b_out  = (const float*)d_in[8];  // [1]
    float* out = (float*)d_out;                   // [365][4096][1]

    char* ws = (char*)d_ws;
    bf16_t* Wcat    = (bf16_t*)ws; ws += (size_t)G4 * KDIM * sizeof(bf16_t);   // 4 MB
    float*  biassum = (float*)ws;  ws += (size_t)G4 * sizeof(float);           // 8 KB
    float*  cbuf    = (float*)ws;  ws += (size_t)NGRID * HID * sizeof(float);  // 8 MB
    bf16_t* x0buf   = (bf16_t*)ws; ws += (size_t)NGRID * HID * sizeof(bf16_t); // 4 MB
    bf16_t* hbuf0   = (bf16_t*)ws; ws += (size_t)NGRID * HID * sizeof(bf16_t); // 4 MB
    bf16_t* hbuf1   = (bf16_t*)ws;                                             // 4 MB

    lstm_init_kernel<<<2048, 256, 0, stream>>>(w_ih, w_hh, b_ih, b_hh,
                                               Wcat, biassum, cbuf, hbuf0, hbuf1);

    bf16_t* hp = hbuf0;
    bf16_t* hn = hbuf1;
    for (int t = 0; t < NTT; ++t) {
        xproj_kernel<<<(NGRID * HID) / 256, 256, 0, stream>>>(
            inputs + (size_t)t * NGRID * NXF, W_in, b_in, x0buf,
            out + (size_t)t * NGRID, b_out);
        lstm_step_kernel<<<dim3(NGRID / 128, HID / 64), 256, 0, stream>>>(
            x0buf, hp, hn, Wcat, biassum, cbuf, W_out, out + (size_t)t * NGRID);
        bf16_t* tmp = hp; hp = hn; hn = tmp;
    }
}
